// LinformerSelfAttention_20547123544322
// MI455X (gfx1250) — matrix operand order repaired and checked by reading, not compiled, not run on hardware
//
#include <hip/hip_runtime.h>
#include <hip/hip_bf16.h>

// ---------------------------------------------------------------------------
// Types for CDNA5 WMMA (wave32): v_wmma_f32_16x16x32_bf16
// ---------------------------------------------------------------------------
typedef __bf16 bf16_t;
typedef __attribute__((ext_vector_type(16))) __bf16 v16bf;
typedef __attribute__((ext_vector_type(8)))  __bf16 v8bf;
typedef __attribute__((ext_vector_type(8)))  float  v8f;
typedef __attribute__((ext_vector_type(4)))  unsigned int v4u;
typedef __attribute__((ext_vector_type(8)))  unsigned int v8u;

#define BB 4
#define NN 4096
#define EE 1024
#define HH 16
#define DD 64
#define KK 256
#define RANK 32

static __device__ __forceinline__ v16bf pack16(v8bf lo, v8bf hi) {
  v16bf r;
#pragma unroll
  for (int i = 0; i < 8; ++i) { r[i] = lo[i]; r[i + 8] = hi[i]; }
  return r;
}

// A operand (16xK tile, row-major, lda elems): lane L holds row L%16,
// K-chunks [kb+8*(L/16), +8) and [kb+16+8*(L/16), +8)  (ISA 16-bit A layout)
static __device__ __forceinline__ v16bf load_a(const bf16_t* A, int lda, int kb, int lane) {
  int m = lane & 15, h = lane >> 4;
  const bf16_t* p = A + (size_t)m * lda + kb + 8 * h;
  v8bf lo = *(const v8bf*)p;
  v8bf hi = *(const v8bf*)(p + 16);
  return pack16(lo, hi);
}

// B operand built from weight W stored (Nout x K) row-major: B[k][n] = W[n][k].
// lane L holds col L%16, K-chunks [kb+8*(L/16), +8) and [kb+16+8*(L/16), +8)
static __device__ __forceinline__ v16bf load_b(const bf16_t* W, int ldw, int kb, int lane) {
  int n = lane & 15, h = lane >> 4;
  const bf16_t* p = W + (size_t)n * ldw + kb + 8 * h;
  v8bf lo = *(const v8bf*)p;
  v8bf hi = *(const v8bf*)(p + 16);
  return pack16(lo, hi);
}

static __device__ __forceinline__ v8f wmma_bf(v16bf a, v16bf b, v8f c) {
  return __builtin_amdgcn_wmma_f32_16x16x32_bf16(false, a, false, b, (short)0, c, false, false);
}

// C layout: lane L, vreg v -> row v + 8*(L/16), col L%16
static __device__ __forceinline__ void store_c_bf16(bf16_t* C, int ldc, v8f c, int lane) {
  int n = lane & 15, h = lane >> 4;
#pragma unroll
  for (int v = 0; v < 8; ++v) C[(size_t)(v + 8 * h) * ldc + n] = (bf16_t)c[v];
}

// ---------------------------------------------------------------------------
// Prep kernels
// ---------------------------------------------------------------------------
__global__ void cvt_f32_bf16(const float* __restrict__ in, bf16_t* __restrict__ out, int n) {
  int i = blockIdx.x * blockDim.x + threadIdx.x;
  int stride = gridDim.x * blockDim.x;
  for (; i < n; i += stride) out[i] = (bf16_t)in[i];
}

// key_proj (4096,256) f32 -> P_T (256,4096) bf16
__global__ void transpose_kp(const float* __restrict__ kp, bf16_t* __restrict__ pt) {
  int i = blockIdx.x * blockDim.x + threadIdx.x;
  if (i >= NN * KK) return;
  int n = i >> 8, k = i & 255;
  pt[(size_t)k * NN + n] = (bf16_t)kp[i];
}

// ---------------------------------------------------------------------------
// K1: qlow = x@qU.T (B*N x 32) ; kT[b][d][n] = (x@kW.T)[b][n][d]
// ---------------------------------------------------------------------------
__global__ __launch_bounds__(256) void k1_qlow_ktrans(
    const bf16_t* __restrict__ xb, const bf16_t* __restrict__ qUb,
    const bf16_t* __restrict__ kWb, bf16_t* __restrict__ qlow,
    bf16_t* __restrict__ kT) {
  int lane = threadIdx.x & 31, wave = threadIdx.x >> 5;
  int mtile = blockIdx.x * 8 + wave;         // 0..1023
  int row0 = mtile * 16;
  v8f acc[6];
#pragma unroll
  for (int t = 0; t < 6; ++t) acc[t] = (v8f){};
  const bf16_t* Ar = xb + (size_t)row0 * EE;
  for (int kb = 0; kb < EE; kb += 32) {
    __builtin_prefetch(Ar + kb + 512, 0, 0);
    v16bf a = load_a(Ar, EE, kb, lane);
#pragma unroll
    for (int t = 0; t < 2; ++t)
      acc[t] = wmma_bf(a, load_b(qUb + (size_t)t * 16 * EE, EE, kb, lane), acc[t]);
#pragma unroll
    for (int t = 0; t < 4; ++t)
      acc[2 + t] = wmma_bf(a, load_b(kWb + (size_t)t * 16 * EE, EE, kb, lane), acc[2 + t]);
  }
  store_c_bf16(qlow + (size_t)row0 * RANK, RANK, acc[0], lane);
  store_c_bf16(qlow + (size_t)row0 * RANK + 16, RANK, acc[1], lane);
  int b = row0 >> 12, nbase = row0 & (NN - 1);
  int n = lane & 15, h = lane >> 4;
#pragma unroll
  for (int t = 0; t < 4; ++t)
#pragma unroll
    for (int v = 0; v < 8; ++v)
      kT[((size_t)b * DD + t * 16 + n) * NN + nbase + v + 8 * h] = (bf16_t)acc[2 + t][v];
}

// ---------------------------------------------------------------------------
// K2: kv[b] (256x64) = P_T (256x4096) x kT[b] (64x4096)^T   (NT WMMA GEMM)
// ---------------------------------------------------------------------------
__global__ __launch_bounds__(32) void k2_kv(const bf16_t* __restrict__ PT,
                                            const bf16_t* __restrict__ kT,
                                            bf16_t* __restrict__ kv) {
  int lane = threadIdx.x & 31;
  int mt = blockIdx.x, nt = blockIdx.y, b = blockIdx.z;
  v8f acc = (v8f){};
  const bf16_t* A = PT + (size_t)mt * 16 * NN;
  const bf16_t* W = kT + ((size_t)b * DD + nt * 16) * NN;
  for (int kb = 0; kb < NN; kb += 32)
    acc = wmma_bf(load_a(A, NN, kb, lane), load_b(W, NN, kb, lane), acc);
  store_c_bf16(kv + ((size_t)b * KK + mt * 16) * DD + nt * 16, DD, acc, lane);
}

// ---------------------------------------------------------------------------
// K3a: q2 = qlow @ qV.T  (16384x32 @ 32x512) — single K-step WMMA
// ---------------------------------------------------------------------------
__global__ __launch_bounds__(256) void k3a_q2(const bf16_t* __restrict__ qlow,
                                              const bf16_t* __restrict__ qVb,
                                              bf16_t* __restrict__ q2) {
  int lane = threadIdx.x & 31, wave = threadIdx.x >> 5;
  int wid = blockIdx.x * 8 + wave;     // 1024 mtiles x 32 ntiles
  int mt = wid >> 5, nt = wid & 31;
  v8f acc = (v8f){};
  acc = wmma_bf(load_a(qlow + (size_t)mt * 16 * RANK, RANK, 0, lane),
                load_b(qVb + (size_t)nt * 16 * RANK, RANK, 0, lane), acc);
  store_c_bf16(q2 + (size_t)mt * 16 * 512 + nt * 16, 512, acc, lane);
}

// ---------------------------------------------------------------------------
// K3b / K5: generic NT GEMM + bias; each wave does one 16x64 tile.
// Waves of one block share the same 16-row A panel (L0/L2 reuse).
// ---------------------------------------------------------------------------
template <int KDIM, int NDIM, bool OUT_F32>
__global__ __launch_bounds__(256) void gemm_nt_bias(const bf16_t* __restrict__ A,
                                                    const bf16_t* __restrict__ W,
                                                    const float* __restrict__ bias,
                                                    void* __restrict__ out) {
  int lane = threadIdx.x & 31, wave = threadIdx.x >> 5;
  int wid = blockIdx.x * 8 + wave;
  const int NG = NDIM / 64;
  int mt = wid / NG, ng = wid % NG;
  v8f acc[4];
#pragma unroll
  for (int t = 0; t < 4; ++t) acc[t] = (v8f){};
  const bf16_t* Ar = A + (size_t)mt * 16 * KDIM;
  for (int kb = 0; kb < KDIM; kb += 32) {
    __builtin_prefetch(Ar + kb + 512, 0, 0);
    v16bf a = load_a(Ar, KDIM, kb, lane);
#pragma unroll
    for (int t = 0; t < 4; ++t)
      acc[t] = wmma_bf(a, load_b(W + (size_t)(ng * 64 + t * 16) * KDIM, KDIM, kb, lane), acc[t]);
  }
  int n = lane & 15, h = lane >> 4;
#pragma unroll
  for (int t = 0; t < 4; ++t) {
    int col = ng * 64 + t * 16 + n;
    float bv = bias[col];
#pragma unroll
    for (int v = 0; v < 8; ++v) {
      size_t off = ((size_t)mt * 16 + v + 8 * h) * NDIM + col;
      if (OUT_F32) ((float*)out)[off] = acc[t][v] + bv;
      else         ((bf16_t*)out)[off] = (bf16_t)(acc[t][v] + bv);
    }
  }
}

// ---------------------------------------------------------------------------
// K4: attention. kv panel (256x64 bf16, 32KB) is DMA'd into LDS by the
// Tensor Data Mover (tensor_load_to_lds + s_wait_tensorcnt), transposed in
// LDS, then: scores via WMMA (B from LDS), softmax via shfl_xor over the
// 16-lane half-groups that own C-matrix rows, w staged through LDS
// (C->A layout), out = w @ kv via WMMA against the LDS-transposed kv.
// ---------------------------------------------------------------------------
__global__ __launch_bounds__(128) void k4_attn(const bf16_t* __restrict__ qf,
                                               const bf16_t* __restrict__ kv,
                                               bf16_t* __restrict__ ao) {
  __shared__ bf16_t kvS[KK * DD];        // 32 KB, TDM destination
  __shared__ bf16_t kvT[DD * KK];        // 32 KB
  __shared__ bf16_t wS[4][16 * KK];      // 32 KB
  int lane = threadIdx.x & 31, wave = threadIdx.x >> 5;
  int bh = blockIdx.x;
  int b = bh >> 4, hh = bh & 15;
  int mt = blockIdx.y * 4 + wave;        // n-tile (of 256 per batch)
  const bf16_t* kvb = kv + (size_t)b * KK * DD;

  // --- TDM: DMA kv[b] (2D tensor 64x256 rows, 2B elems) into kvS ---------
  if (threadIdx.x < 32) {
    unsigned long long ga = (unsigned long long)(uintptr_t)kvb;   // global byte addr
    unsigned int lds_off = (unsigned int)(uintptr_t)(&kvS[0]);    // LDS aperture: low 32 = offset
    v4u g0;
    g0[0] = 1u;                                           // count=1, user descriptor
    g0[1] = lds_off;                                      // lds_addr
    g0[2] = (unsigned int)(ga & 0xffffffffu);             // global_addr[31:0]
    g0[3] = (unsigned int)((ga >> 32) & 0x1ffffffu) | (2u << 30); // addr[56:32] | type=2
    v8u g1;
    g1[0] = 1u << 16;                                     // data_size = 2 bytes
    g1[1] = ((unsigned int)DD) << 16;                     // tensor_dim0 = 64
    g1[2] = ((unsigned int)KK) << 16;                     // tensor_dim1 = 256
    g1[3] = ((unsigned int)DD) << 16;                     // tile_dim0 = 64
    g1[4] = (unsigned int)KK;                             // tile_dim1 = 256, tile_dim2 = 0
    g1[5] = (unsigned int)DD;                             // tensor_dim0_stride = 64
    g1[6] = 0u;
    g1[7] = 0u;
    asm volatile("tensor_load_to_lds %0, %1" :: "s"(g0), "s"(g1) : "memory");
    __builtin_amdgcn_s_wait_tensorcnt(0);
  }
  __syncthreads();

  // transpose kvS -> kvT (LDS to LDS)
  for (int i = threadIdx.x; i < KK * DD; i += 128) {
    int k = i >> 6, d = i & 63;
    kvT[d * KK + k] = kvS[i];
  }
  __syncthreads();

  const bf16_t* qrow = qf + ((size_t)b * NN + mt * 16) * EE + hh * DD;
  v8f sc[16];
#pragma unroll
  for (int t = 0; t < 16; ++t) sc[t] = (v8f){};
  for (int kb = 0; kb < DD; kb += 32) {
    v16bf a = load_a(qrow, EE, kb, lane);
#pragma unroll
    for (int t = 0; t < 16; ++t)
      sc[t] = wmma_bf(a, load_b(kvS + (size_t)t * 16 * DD, DD, kb, lane), sc[t]);
  }

  const float scale = 0.125f;            // 1/sqrt(64)
  int n = lane & 15, h = lane >> 4;
#pragma unroll
  for (int v = 0; v < 8; ++v) {
    float mx = -3.0e38f;
#pragma unroll
    for (int t = 0; t < 16; ++t) mx = fmaxf(mx, sc[t][v]);
    mx = fmaxf(mx, __shfl_xor(mx, 1));
    mx = fmaxf(mx, __shfl_xor(mx, 2));
    mx = fmaxf(mx, __shfl_xor(mx, 4));
    mx = fmaxf(mx, __shfl_xor(mx, 8));
    float s = 0.f;
#pragma unroll
    for (int t = 0; t < 16; ++t) {
      float e = __expf((sc[t][v] - mx) * scale);
      sc[t][v] = e; s += e;
    }
    s += __shfl_xor(s, 1);
    s += __shfl_xor(s, 2);
    s += __shfl_xor(s, 4);
    s += __shfl_xor(s, 8);
    float inv = 1.f / s;
#pragma unroll
    for (int t = 0; t < 16; ++t) sc[t][v] *= inv;
  }

  bf16_t* wrow = wS[wave];
#pragma unroll
  for (int t = 0; t < 16; ++t)
#pragma unroll
    for (int v = 0; v < 8; ++v)
      wrow[(v + 8 * h) * KK + t * 16 + n] = (bf16_t)sc[t][v];
  asm volatile("s_wait_dscnt 0" ::: "memory");

  v8f o[4];
#pragma unroll
  for (int t = 0; t < 4; ++t) o[t] = (v8f){};
  for (int kb = 0; kb < KK; kb += 32) {
    v16bf a = load_a(wrow, KK, kb, lane);
#pragma unroll
    for (int t = 0; t < 4; ++t)
      o[t] = wmma_bf(a, load_b(kvT + (size_t)t * 16 * KK, KK, kb, lane), o[t]);
  }
  bf16_t* orow = ao + ((size_t)b * NN + mt * 16) * EE + hh * DD;
#pragma unroll
  for (int t = 0; t < 4; ++t) store_c_bf16(orow + t * 16, EE, o[t], lane);
}

// ---------------------------------------------------------------------------
// Host side
// ---------------------------------------------------------------------------
extern "C" void kernel_launch(void* const* d_in, const int* in_sizes, int n_in,
                              void* d_out, int out_size, void* d_ws, size_t ws_size,
                              hipStream_t stream) {
  const float* x  = (const float*)d_in[0];
  const float* qU = (const float*)d_in[1];
  const float* qV = (const float*)d_in[2];
  const float* qW = (const float*)d_in[3];
  const float* qb = (const float*)d_in[4];
  const float* kW = (const float*)d_in[5];
  const float* kp = (const float*)d_in[6];
  const float* oW = (const float*)d_in[7];
  const float* ob = (const float*)d_in[8];
  float* out = (float*)d_out;

  char* ws = (char*)d_ws;
  size_t off = 0;
  auto alloc = [&](size_t bytes) { char* p = ws + off; off += (bytes + 255) & ~(size_t)255; return p; };
  bf16_t* xb    = (bf16_t*)alloc((size_t)BB * NN * EE * 2);     // also reused for attn out
  bf16_t* qUb   = (bf16_t*)alloc((size_t)RANK * EE * 2);
  bf16_t* qVb   = (bf16_t*)alloc((size_t)512 * RANK * 2);
  bf16_t* qWb   = (bf16_t*)alloc((size_t)1024 * 512 * 2);
  bf16_t* kWb   = (bf16_t*)alloc((size_t)DD * EE * 2);
  bf16_t* oWb   = (bf16_t*)alloc((size_t)EE * 1024 * 2);
  bf16_t* PT    = (bf16_t*)alloc((size_t)KK * NN * 2);
  bf16_t* qlow  = (bf16_t*)alloc((size_t)BB * NN * RANK * 2);
  bf16_t* kT    = (bf16_t*)alloc((size_t)BB * DD * NN * 2);
  bf16_t* kvb   = (bf16_t*)alloc((size_t)BB * KK * DD * 2);
  bf16_t* q2    = (bf16_t*)alloc((size_t)BB * NN * 512 * 2);
  bf16_t* qfull = (bf16_t*)alloc((size_t)BB * NN * EE * 2);
  bf16_t* attn  = xb;  // alias: x_bf dead after K1/K3

  auto cvt = [&](const float* src, bf16_t* dst, int n) {
    cvt_f32_bf16<<<(n + 255) / 256, 256, 0, stream>>>(src, dst, n);
  };
  cvt(x,  xb,  BB * NN * EE);
  cvt(qU, qUb, RANK * EE);
  cvt(qV, qVb, 512 * RANK);
  cvt(qW, qWb, 1024 * 512);
  cvt(kW, kWb, DD * EE);
  cvt(oW, oWb, EE * 1024);
  transpose_kp<<<(NN * KK + 255) / 256, 256, 0, stream>>>(kp, PT);

  k1_qlow_ktrans<<<128, 256, 0, stream>>>(xb, qUb, kWb, qlow, kT);
  k2_kv<<<dim3(16, 4, 4), 32, 0, stream>>>(PT, kT, kvb);
  k3a_q2<<<4096, 256, 0, stream>>>(qlow, qVb, q2);
  gemm_nt_bias<512, 1024, false><<<2048, 256, 0, stream>>>(q2, qWb, qb, qfull);
  k4_attn<<<dim3(BB * HH, NN / 64), 128, 0, stream>>>(qfull, kvb, attn);
  gemm_nt_bias<1024, 1024, true><<<2048, 256, 0, stream>>>(attn, oWb, ob, out);
  (void)in_sizes; (void)n_in; (void)out_size; (void)ws_size;
}
